// GroupQueryAttention_38938173506113
// MI455X (gfx1250) — matrix operand order repaired and checked by reading, not compiled, not run on hardware
//
#include <hip/hip_runtime.h>
#include <hip/hip_bf16.h>

typedef __attribute__((ext_vector_type(16))) __bf16 v16bf;
typedef __attribute__((ext_vector_type(8)))  float  v8f;

#define B_ 2
#define S_ 2048
#define E_ 2048
#define H_ 32
#define G_ 8
#define D_ 64
#define HD_ (H_*D_)   // 2048
#define GD_ (G_*D_)   // 512
#define M_  (B_*S_)   // 4096

// ---------------------------------------------------------------------------
// CDNA5 async copy: 16 bytes global -> LDS, tracked by ASYNCcnt.
// LDS operand = low 32 bits of the flat address (LDS aperture: addr[31:0]).
// ---------------------------------------------------------------------------
__device__ __forceinline__ void async_cp16(void* lds_dst, const void* gsrc)
{
    unsigned off = (unsigned)(unsigned long long)lds_dst;
    asm volatile("global_load_async_to_lds_b128 %0, %1, off"
                 :: "v"(off), "v"(gsrc) : "memory");
}
#define WAIT_ASYNC() asm volatile("s_wait_asynccnt 0x0" ::: "memory")
#define WAIT_DS()    asm volatile("s_wait_dscnt 0x0"    ::: "memory")

// ---------------------------------------------------------------------------
// GEMM: C[4096,N] = A[4096,K](bf16) * Bt[N,K](bf16, pre-transposed weights).
// Compile-time K,N. Block = 256 thr = 8 waves (4M x 2N), tile 128x128.
// Wave tile 32x64 -> 8 WMMAs / 32-K step. Double-buffered async LDS staging.
// ---------------------------------------------------------------------------
template<int K, int N>
__global__ __launch_bounds__(256)
void gemm_bf16wmma_kernel(const __bf16* __restrict__ A, const __bf16* __restrict__ Bt,
                          float* __restrict__ C)
{
    __shared__ __bf16 As[2][128][40];   // [m][k], 80B rows (16B aligned)
    __shared__ __bf16 Bs[2][128][40];   // [n][k]

    const int tid  = threadIdx.x;
    const int lane = tid & 31;
    const int wave = tid >> 5;
    const int wm   = wave >> 1;          // 0..3
    const int wn   = wave & 1;           // 0..1
    const int half = lane >> 4;
    const int ml   = lane & 15;
    const int mBlk = blockIdx.y * 128;
    const int nBlk = blockIdx.x * 128;

    v8f acc[2][4];
#pragma unroll
    for (int am = 0; am < 2; ++am)
#pragma unroll
        for (int nt = 0; nt < 4; ++nt) {
            v8f z = {0.f,0.f,0.f,0.f,0.f,0.f,0.f,0.f};
            acc[am][nt] = z;
        }

    // stage one 128x32 A tile + 128x32 Bt tile: 4 x 16B async per thread
    auto stage = [&](int buf, int k0) {
#pragma unroll
        for (int i = 0; i < 2; ++i) {
            int idx = tid + 256 * i;            // 0..511
            int r = idx >> 2, c = idx & 3;
            async_cp16(&As[buf][r][c * 8], &A [(size_t)(mBlk + r) * K + k0 + c * 8]);
            async_cp16(&Bs[buf][r][c * 8], &Bt[(size_t)(nBlk + r) * K + k0 + c * 8]);
        }
    };

    stage(0, 0);
    int cur = 0;
    for (int k0 = 0; k0 < K; k0 += 32) {
        WAIT_DS();            // my fragment reads of buffer cur^1 retired
        WAIT_ASYNC();         // my async copies into buffer cur landed
        __syncthreads();      // everyone's copies visible / reads done
        if (k0 + 32 < K) stage(cur ^ 1, k0 + 32);

        // A fragments: 16x32, lane=m, k = 16*half + e
        v16bf a[2];
#pragma unroll
        for (int am = 0; am < 2; ++am)
#pragma unroll
            for (int e = 0; e < 16; ++e)
                a[am][e] = As[cur][wm * 32 + am * 16 + ml][16 * half + e];

#pragma unroll
        for (int nt = 0; nt < 4; ++nt) {
            v16bf b;                            // 32x16, lane=n, k = 16*half + e
#pragma unroll
            for (int e = 0; e < 16; ++e)
                b[e] = Bs[cur][wn * 64 + nt * 16 + ml][16 * half + e];
#pragma unroll
            for (int am = 0; am < 2; ++am)
                acc[am][nt] = __builtin_amdgcn_wmma_f32_16x16x32_bf16(
                    false, a[am], false, b, (short)0, acc[am][nt], false, false);
        }
        cur ^= 1;
    }

#pragma unroll
    for (int am = 0; am < 2; ++am)
#pragma unroll
        for (int nt = 0; nt < 4; ++nt)
#pragma unroll
            for (int v = 0; v < 8; ++v)
                C[(size_t)(mBlk + wm * 32 + am * 16 + v + 8 * half) * N +
                  (nBlk + wn * 64 + nt * 16 + ml)] = acc[am][nt][v];
}

// fp32 -> bf16 elementwise
__global__ __launch_bounds__(256)
void cvt_kernel(const float* __restrict__ src, __bf16* __restrict__ dst)
{
    const int i = blockIdx.x * 256 + threadIdx.x;
    dst[i] = (__bf16)src[i];
}

// fp32 [R][C] -> bf16 [C][R] (transposed weights), 32x32 LDS tile
__global__ __launch_bounds__(256)
void cvtT_kernel(const float* __restrict__ src, __bf16* __restrict__ dst, int R, int Cc)
{
    __shared__ float t[32][33];
    const int lx = threadIdx.x & 31, ly = threadIdx.x >> 5;
#pragma unroll
    for (int i = 0; i < 4; ++i)
        t[ly + i * 8][lx] = src[(size_t)(blockIdx.y * 32 + ly + i * 8) * Cc + blockIdx.x * 32 + lx];
    __syncthreads();
#pragma unroll
    for (int i = 0; i < 4; ++i)
        dst[(size_t)(blockIdx.x * 32 + ly + i * 8) * R + blockIdx.y * 32 + lx] =
            (__bf16)t[lx][ly + i * 8];
}

// ---------------------------------------------------------------------------
// RMSNorm + RoPE: one wave32 per (b,s,head) D=64 vector; lane = RoPE pair.
// Output bf16, head-major [b, head, s, d].
// ---------------------------------------------------------------------------
__global__ __launch_bounds__(256)
void normrope_kernel(const float* __restrict__ src, __bf16* __restrict__ dst,
                     const float* __restrict__ gamma, const int* __restrict__ pos_ids,
                     int nheads)
{
    const int lane = threadIdx.x & 31;
    const int wave = threadIdx.x >> 5;
    const int vec  = blockIdx.x * 8 + wave;          // (b*S + s)*nheads + h
    const int h  = vec % nheads;
    const int bs = vec / nheads;
    const int s  = bs % S_;
    const int b  = bs / S_;

    const float x1 = src[(size_t)vec * D_ + lane];
    const float x2 = src[(size_t)vec * D_ + lane + 32];
    float ss = x1 * x1 + x2 * x2;
#pragma unroll
    for (int o = 16; o >= 1; o >>= 1) ss += __shfl_xor(ss, o, 32);
    const float rms = sqrtf(ss * (1.0f / 64.0f));
    const float inv = 1.0f / (rms + 1e-6f);
    const float n1 = x1 * inv * gamma[lane];
    const float n2 = x2 * inv * gamma[lane + 32];

    const float pos  = (float)pos_ids[b * S_ + s];
    const float freq = __powf(10000.0f, -(float)(2 * lane) * (1.0f / 64.0f));
    const float ang  = pos * freq;
    const float c = cosf(ang), sn = sinf(ang);

    __bf16* o = dst + ((size_t)(b * nheads + h) * S_ + s) * D_;
    o[lane]      = (__bf16)(n1 * c - n2 * sn);
    o[lane + 32] = (__bf16)(n2 * c + n1 * sn);
}

// V: fp32 [b,s,g,d] -> bf16 [b,g,d,s]  (d-major so attention V tiles are
// contiguous 16B runs along s for async staging)
__global__ __launch_bounds__(256)
void v_transpose_kernel(const float* __restrict__ src, __bf16* __restrict__ dst)
{
    const int idx = blockIdx.x * 256 + threadIdx.x;
    const int d = idx & 63;
    const int g = (idx >> 6) & 7;
    const int s = (idx >> 9) & 2047;
    const int b = idx >> 20;
    dst[((size_t)(b * G_ + g) * D_ + d) * S_ + s] = (__bf16)src[idx];
}

// ---------------------------------------------------------------------------
// Causal GQA flash attention. Block = 128 thr = 4 waves; wave owns 16 queries.
// Double-buffered async staging of K[32][64] and V^T[64][32] tiles.
// Scores + P*V via bf16 WMMA; fp32 online softmax; bf16 context out.
// ---------------------------------------------------------------------------
__global__ __launch_bounds__(128)
void gqa_flash_kernel(const __bf16* __restrict__ Qbf, const __bf16* __restrict__ Kbf,
                      const __bf16* __restrict__ Vt, __bf16* __restrict__ ctx)
{
    __shared__ __bf16 Ks[2][32][72];    // [key][d], 144B rows
    __shared__ __bf16 Vs[2][64][40];    // [d][key], 80B rows
    __shared__ __bf16 Ps[4][16][40];    // per-wave P relayout

    const int tid  = threadIdx.x;
    const int lane = tid & 31;
    const int w    = tid >> 5;
    const int half = lane >> 4;
    const int ml   = lane & 15;

    const int bh = blockIdx.y;          // b*H + h
    const int b  = bh / H_;
    const int h  = bh % H_;
    const int g  = h / (H_ / G_);
    const int qb0   = blockIdx.x * 64;
    const int qrow0 = qb0 + w * 16;

    const __bf16* Qp  = Qbf + (size_t)bh * S_ * D_;
    const __bf16* Kp  = Kbf + (size_t)(b * G_ + g) * S_ * D_;
    const __bf16* Vtp = Vt  + (size_t)(b * G_ + g) * D_ * S_;

    // Q fragments held in registers (A = Q[16,64])
    v16bf aq0, aq1;
    {
        const __bf16* qr = Qp + (size_t)(qrow0 + ml) * D_;
#pragma unroll
        for (int e = 0; e < 16; ++e) {
            const int kp = 16 * half + e;
            aq0[e] = qr[kp];
            aq1[e] = qr[32 + kp];
        }
    }

    v8f acc[4];
#pragma unroll
    for (int i = 0; i < 4; ++i) {
        v8f z = {0.f,0.f,0.f,0.f,0.f,0.f,0.f,0.f};
        acc[i] = z;
    }
    float m8[8], l8[8];
#pragma unroll
    for (int v = 0; v < 8; ++v) { m8[v] = -1e30f; l8[v] = 0.0f; }

    const int   nkb   = (qb0 + 64) / 32;
    const int   qmax  = qrow0 + 15;
    const float scale = 0.125f;

    // async-stage one 32-key block (K rows 128B, Vt rows 64B)
    auto stage = [&](int buf, int kb) {
#pragma unroll
        for (int i = 0; i < 2; ++i) {
            int idx = tid + 128 * i;                     // 0..255
            int kr = idx >> 3, kc = idx & 7;             // 32 x 8 chunks
            async_cp16(&Ks[buf][kr][kc * 8], &Kp[(size_t)(kb * 32 + kr) * D_ + kc * 8]);
            int vr = idx >> 2, vc = idx & 3;             // 64 x 4 chunks
            async_cp16(&Vs[buf][vr][vc * 8], &Vtp[(size_t)vr * S_ + kb * 32 + vc * 8]);
        }
    };

    stage(0, 0);
    int cur = 0;
    for (int kb = 0; kb < nkb; ++kb) {
        WAIT_DS();
        WAIT_ASYNC();
        __syncthreads();
        if (kb + 1 < nkb) stage(cur ^ 1, kb + 1);

        if (kb * 32 <= qmax) {
            // scores: two 16x16 tiles (keys 0-15, 16-31), 2 WMMAs each over d=64
            v8f sc[2];
#pragma unroll
            for (int kt = 0; kt < 2; ++kt) {
                v8f z = {0.f,0.f,0.f,0.f,0.f,0.f,0.f,0.f};
                sc[kt] = z;
                v16bf b0, b1;
#pragma unroll
                for (int e = 0; e < 16; ++e) {
                    b0[e] = Ks[cur][kt * 16 + ml][16 * half + e];
                    b1[e] = Ks[cur][kt * 16 + ml][32 + 16 * half + e];
                }
                sc[kt] = __builtin_amdgcn_wmma_f32_16x16x32_bf16(
                    false, aq0, false, b0, (short)0, sc[kt], false, false);
                sc[kt] = __builtin_amdgcn_wmma_f32_16x16x32_bf16(
                    false, aq1, false, b1, (short)0, sc[kt], false, false);
            }

            // online softmax (row = v + 8*half; 16-lane half-group reductions)
            const int key0 = kb * 32 + ml;
            const int key1 = key0 + 16;
#pragma unroll
            for (int v = 0; v < 8; ++v) {
                const int q = qrow0 + v + 8 * half;
                float s0 = (key0 <= q) ? sc[0][v] * scale : -1e30f;
                float s1 = (key1 <= q) ? sc[1][v] * scale : -1e30f;
                float mx = fmaxf(s0, s1);
#pragma unroll
                for (int o = 8; o >= 1; o >>= 1) mx = fmaxf(mx, __shfl_xor(mx, o, 32));
                const float mnew  = fmaxf(m8[v], mx);
                const float alpha = __expf(m8[v] - mnew);
                const float p0 = (key0 <= q) ? __expf(s0 - mnew) : 0.0f;
                const float p1 = (key1 <= q) ? __expf(s1 - mnew) : 0.0f;
                float psum = p0 + p1;
#pragma unroll
                for (int o = 8; o >= 1; o >>= 1) psum += __shfl_xor(psum, o, 32);
                l8[v] = l8[v] * alpha + psum;
                m8[v] = mnew;
#pragma unroll
                for (int nt = 0; nt < 4; ++nt) acc[nt][v] *= alpha;
                Ps[w][v + 8 * half][ml]      = (__bf16)p0;
                Ps[w][v + 8 * half][16 + ml] = (__bf16)p1;
            }
            __builtin_amdgcn_wave_barrier();     // LDS relayout within wave

            // P fragment (A layout 16x32) and P*V (4 WMMAs over d tiles)
            v16bf ap;
#pragma unroll
            for (int e = 0; e < 16; ++e)
                ap[e] = Ps[w][ml][16 * half + e];
#pragma unroll
            for (int nt = 0; nt < 4; ++nt) {
                v16bf bv;
#pragma unroll
                for (int e = 0; e < 16; ++e)
                    bv[e] = Vs[cur][nt * 16 + ml][16 * half + e];
                acc[nt] = __builtin_amdgcn_wmma_f32_16x16x32_bf16(
                    false, ap, false, bv, (short)0, acc[nt], false, false);
            }
        }
        cur ^= 1;
    }

    // normalize, write bf16 context [b, s, h, d] (A-matrix of output GEMM)
#pragma unroll
    for (int v = 0; v < 8; ++v) {
        const int q = qrow0 + v + 8 * half;
        const float invl = 1.0f / l8[v];
#pragma unroll
        for (int nt = 0; nt < 4; ++nt)
            ctx[((size_t)(b * S_ + q) * H_ + h) * D_ + nt * 16 + ml] =
                (__bf16)(acc[nt][v] * invl);
    }
}

// ---------------------------------------------------------------------------
extern "C" void kernel_launch(void* const* d_in, const int* in_sizes, int n_in,
                              void* d_out, int out_size, void* d_ws, size_t ws_size,
                              hipStream_t stream)
{
    (void)in_sizes; (void)n_in; (void)out_size; (void)ws_size;
    const float* x  = (const float*)d_in[0];
    const float* Wq = (const float*)d_in[1];
    const float* Wk = (const float*)d_in[2];
    const float* Wv = (const float*)d_in[3];
    const float* Wo = (const float*)d_in[4];
    const float* qg = (const float*)d_in[5];
    const float* kg = (const float*)d_in[6];
    const int*  pos = (const int*)d_in[7];
    float* out = (float*)d_out;

    char* ws = (char*)d_ws;
    __bf16* xbf  = (__bf16*)ws; ws += (size_t)M_ * E_  * 2;
    __bf16* Wqt  = (__bf16*)ws; ws += (size_t)E_ * HD_ * 2;
    __bf16* Wkt  = (__bf16*)ws; ws += (size_t)E_ * GD_ * 2;
    __bf16* Wvt  = (__bf16*)ws; ws += (size_t)E_ * GD_ * 2;
    __bf16* Wot  = (__bf16*)ws; ws += (size_t)HD_ * E_ * 2;
    float*  Qraw = (float*)ws;  ws += (size_t)M_ * HD_ * 4;
    float*  Kraw = (float*)ws;  ws += (size_t)M_ * GD_ * 4;
    float*  Vraw = (float*)ws;  ws += (size_t)M_ * GD_ * 4;
    __bf16* Qbf  = (__bf16*)ws; ws += (size_t)M_ * HD_ * 2;
    __bf16* Kbf  = (__bf16*)ws; ws += (size_t)M_ * GD_ * 2;
    __bf16* Vtbf = (__bf16*)ws; ws += (size_t)M_ * GD_ * 2;
    __bf16* ctx  = (__bf16*)ws; ws += (size_t)M_ * HD_ * 2;

    // one-time bf16 conversion / weight transposes (all L2-resident afterwards)
    cvt_kernel <<<(M_*E_)/256, 256, 0, stream>>>(x, xbf);
    cvtT_kernel<<<dim3(HD_/32, E_/32), 256, 0, stream>>>(Wq, Wqt, E_, HD_);
    cvtT_kernel<<<dim3(GD_/32, E_/32), 256, 0, stream>>>(Wk, Wkt, E_, GD_);
    cvtT_kernel<<<dim3(GD_/32, E_/32), 256, 0, stream>>>(Wv, Wvt, E_, GD_);
    cvtT_kernel<<<dim3(E_/32, HD_/32), 256, 0, stream>>>(Wo, Wot, HD_, E_);

    // QKV projections
    gemm_bf16wmma_kernel<E_, HD_><<<dim3(HD_/128, M_/128), 256, 0, stream>>>(xbf, Wqt, Qraw);
    gemm_bf16wmma_kernel<E_, GD_><<<dim3(GD_/128, M_/128), 256, 0, stream>>>(xbf, Wkt, Kraw);
    gemm_bf16wmma_kernel<E_, GD_><<<dim3(GD_/128, M_/128), 256, 0, stream>>>(xbf, Wvt, Vraw);

    // RMSNorm + RoPE (Q, K), V relayout
    normrope_kernel<<<(B_*S_*H_)/8, 256, 0, stream>>>(Qraw, Qbf, qg, pos, H_);
    normrope_kernel<<<(B_*S_*G_)/8, 256, 0, stream>>>(Kraw, Kbf, kg, pos, G_);
    v_transpose_kernel<<<(M_*GD_)/256, 256, 0, stream>>>(Vraw, Vtbf);

    // causal GQA flash attention
    gqa_flash_kernel<<<dim3(S_/64, B_*H_), 128, 0, stream>>>(Qbf, Kbf, Vtbf, ctx);

    // output projection
    gemm_bf16wmma_kernel<HD_, E_><<<dim3(E_/128, M_/128), 256, 0, stream>>>(ctx, Wot, out);
}
